// GraphConv_31052613550316
// MI455X (gfx1250) — matrix operand order repaired and checked by reading, not compiled, not run on hardware
//
#include <hip/hip_runtime.h>

typedef __attribute__((ext_vector_type(16))) _Float16 v16h;
typedef __attribute__((ext_vector_type(8)))  float    v8f;

#define IN_F  512
#define RANK  64
#define OUT_F 128

// Branchless fast tanh: (e^{2x}-1) * rcp(e^{2x}+1), clamped so e stays finite.
__device__ __forceinline__ float fast_tanhf(float x) {
    float xc = fminf(fmaxf(x, -15.0f), 15.0f);
    float t  = __expf(2.0f * xc);
    return (t - 1.0f) * __builtin_amdgcn_rcpf(t + 1.0f);
}

// ---------------------------------------------------------------------------
// Init: zero accumulators/degrees; pre-shuffle weights into WMMA B-fragment-
// major f16 layout so GEMMs load each 32-byte fragment with one v16h load.
// Fragment order: ((ks*NT + nt)*32 + lane)*16 + e, where K = ks*32 + kbB + e + (e<8 ? 0 : 8),
// kbB = (lane<16 ? 0 : 8), N = nt*16 + (lane&15).
// ---------------------------------------------------------------------------
__global__ void k_init(int* __restrict__ outdeg, int* __restrict__ indeg,
                       float* __restrict__ logs, unsigned* __restrict__ sgn,
                       const float* __restrict__ w1, const float* __restrict__ w2,
                       _Float16* __restrict__ w1frag, _Float16* __restrict__ w2frag,
                       int nN) {
    int i = blockIdx.x * blockDim.x + threadIdx.x;
    int NR = nN * RANK;
    if (i < NR) { logs[i] = 0.0f; sgn[i] = 0u; }
    if (i < nN) { outdeg[i] = 0; indeg[i] = 0; }

    if (i < IN_F * RANK) {              // 16 ksteps x 4 ntiles x 32 lanes x 16
        int e    = i & 15;
        int lane = (i >> 4) & 31;
        int nt   = (i >> 9) & 3;
        int ks   = i >> 11;
        int kbB  = (lane < 16) ? 0 : 8;
        int K    = ks * 32 + kbB + e + ((e < 8) ? 0 : 8);
        int N    = nt * 16 + (lane & 15);
        w1frag[i] = (_Float16)w1[K * RANK + N];
    }
    if (i < RANK * OUT_F) {             // 2 ksteps x 8 ntiles x 32 lanes x 16
        int e    = i & 15;
        int lane = (i >> 4) & 31;
        int nt   = (i >> 9) & 7;
        int ks   = i >> 12;
        int kbB  = (lane < 16) ? 0 : 8;
        int K    = ks * 32 + kbB + e + ((e < 8) ? 0 : 8);
        int N    = nt * 16 + (lane & 15);
        w2frag[i] = (_Float16)w2[K * OUT_F + N];
    }
}

// ---------------------------------------------------------------------------
// Degree counting (segment_sum of ones)
// ---------------------------------------------------------------------------
__global__ void k_deg(const int* __restrict__ src, const int* __restrict__ dst,
                      int* __restrict__ outdeg, int* __restrict__ indeg, int nE) {
    int e = blockIdx.x * blockDim.x + threadIdx.x;
    if (e < nE) {
        atomicAdd(&outdeg[src[e]], 1);
        atomicAdd(&indeg[dst[e]], 1);
    }
}

// ---------------------------------------------------------------------------
// GEMM1: h = tanh((feat * out_norm) @ W)   [nN x 512] @ [512 x 64]
// One wave computes a 16-row x 64-col tile via 4 WMMA accumulators.
// ---------------------------------------------------------------------------
__global__ __launch_bounds__(256)
void k_gemm1(const float* __restrict__ feat, const _Float16* __restrict__ w1frag,
             const int* __restrict__ outdeg, float* __restrict__ h, int mtiles) {
    int wave = (blockIdx.x * blockDim.x + threadIdx.x) >> 5;
    if (wave >= mtiles) return;               // wave-uniform: EXEC stays all-ones
    int lane = threadIdx.x & 31;
    int m0   = wave * 16;
    int row  = m0 + (lane & 15);
    int koff = (lane < 16) ? 0 : 8;           // A-frag K sub-block per ISA layout
    int ncol = lane & 15;

    float norm = rsqrtf(fmaxf((float)outdeg[row], 1.0f));

    v8f acc[4] = {};
    const float* fr = feat + (size_t)row * IN_F;
    const v16h*  bf = (const v16h*)w1frag;

    for (int ks = 0; ks < IN_F / 32; ++ks) {
        int kbase = ks * 32;
        __builtin_prefetch(fr + kbase + 64, 0, 1);   // global_prefetch_b8
        // A fragment: two contiguous 8-float runs -> four b128 loads + cvt
        float4 q0 = *(const float4*)(fr + kbase + koff);
        float4 q1 = *(const float4*)(fr + kbase + koff + 4);
        float4 q2 = *(const float4*)(fr + kbase + 16 + koff);
        float4 q3 = *(const float4*)(fr + kbase + 16 + koff + 4);
        v16h a;
        a[0]  = (_Float16)(q0.x * norm); a[1]  = (_Float16)(q0.y * norm);
        a[2]  = (_Float16)(q0.z * norm); a[3]  = (_Float16)(q0.w * norm);
        a[4]  = (_Float16)(q1.x * norm); a[5]  = (_Float16)(q1.y * norm);
        a[6]  = (_Float16)(q1.z * norm); a[7]  = (_Float16)(q1.w * norm);
        a[8]  = (_Float16)(q2.x * norm); a[9]  = (_Float16)(q2.y * norm);
        a[10] = (_Float16)(q2.z * norm); a[11] = (_Float16)(q2.w * norm);
        a[12] = (_Float16)(q3.x * norm); a[13] = (_Float16)(q3.y * norm);
        a[14] = (_Float16)(q3.z * norm); a[15] = (_Float16)(q3.w * norm);

        #pragma unroll
        for (int nt = 0; nt < 4; ++nt) {
            v16h b = bf[(ks * 4 + nt) * 32 + lane];  // one 32B fragment load
            acc[nt] = __builtin_amdgcn_wmma_f32_16x16x32_f16(
                false, a, false, b, (short)0, acc[nt], false, false);
        }
    }
    // Store tanh(acc). D layout: VGPR r -> row m0 + r + 8*(lane>=16), col = lane&15
    int rbase = m0 + ((lane < 16) ? 0 : 8);
    #pragma unroll
    for (int nt = 0; nt < 4; ++nt)
        #pragma unroll
        for (int r = 0; r < 8; ++r)
            h[(size_t)(rbase + r) * RANK + nt * 16 + ncol] = fast_tanhf(acc[nt][r]);
}

// ---------------------------------------------------------------------------
// Edge scatter: segment_prod in log space, 4 ranks per thread.
//   m = h[src][r] * ew[e];  logsum += log|m| (log(0) = -inf => product 0);
//   sign parity via atomicXor. Accumulators are L2-resident (25.6 MB each).
// ---------------------------------------------------------------------------
__global__ void k_edge(const float* __restrict__ h, const float* __restrict__ ew,
                       const int* __restrict__ src, const int* __restrict__ dst,
                       float* __restrict__ logs, unsigned* __restrict__ sgn,
                       long long total) {
    long long tid = (long long)blockIdx.x * blockDim.x + threadIdx.x;
    if (tid >= total) return;
    int e  = (int)(tid >> 4);
    int r4 = (int)(tid & 15) * 4;
    int s = src[e], d = dst[e];
    float w = ew[e];
    float4 hv = *(const float4*)(h + (size_t)s * RANK + r4);
    size_t o = (size_t)d * RANK + r4;
    float m0 = hv.x * w, m1 = hv.y * w, m2 = hv.z * w, m3 = hv.w * w;
    atomicAdd(&logs[o + 0], __logf(fabsf(m0)));
    atomicAdd(&logs[o + 1], __logf(fabsf(m1)));
    atomicAdd(&logs[o + 2], __logf(fabsf(m2)));
    atomicAdd(&logs[o + 3], __logf(fabsf(m3)));
    if (m0 < 0.0f) atomicXor(&sgn[o + 0], 1u);
    if (m1 < 0.0f) atomicXor(&sgn[o + 1], 1u);
    if (m2 < 0.0f) atomicXor(&sgn[o + 2], 1u);
    if (m3 < 0.0f) atomicXor(&sgn[o + 3], 1u);
}

// ---------------------------------------------------------------------------
// GEMM2 (fused finalize, branchless):
//   g = select(in_deg>0, sign*exp(logsum), h) * in_norm ;  out = g @ W2
//   [nN x 64] @ [64 x 128], one wave = 16 rows x 128 cols (8 accumulators)
// ---------------------------------------------------------------------------
__device__ __forceinline__ float g_elem(float lg, unsigned sg, float hv,
                                        bool useAgg, float norm) {
    float e = __expf(lg);
    float a = (sg & 1u) ? -e : e;
    return (useAgg ? a : hv) * norm;
}

__global__ __launch_bounds__(256)
void k_gemm2(const float* __restrict__ logs, const unsigned* __restrict__ sgn,
             const float* __restrict__ h, const int* __restrict__ indeg,
             const _Float16* __restrict__ w2frag, float* __restrict__ out, int mtiles) {
    int wave = (blockIdx.x * blockDim.x + threadIdx.x) >> 5;
    if (wave >= mtiles) return;
    int lane = threadIdx.x & 31;
    int m0   = wave * 16;
    int row  = m0 + (lane & 15);
    int koff = (lane < 16) ? 0 : 8;
    int ncol = lane & 15;

    float ind   = (float)indeg[row];
    float norm  = rsqrtf(fmaxf(ind, 1.0f));
    bool useAgg = (ind > 0.0f);

    v8f acc[8] = {};
    const v16h* bf = (const v16h*)w2frag;

    for (int ks = 0; ks < RANK / 32; ++ks) {
        size_t i0 = (size_t)row * RANK + ks * 32 + koff;       // 8 contiguous
        size_t i1 = i0 + 16;                                   // 8 contiguous
        float4 l0 = *(const float4*)(logs + i0), l1 = *(const float4*)(logs + i0 + 4);
        float4 l2 = *(const float4*)(logs + i1), l3 = *(const float4*)(logs + i1 + 4);
        uint4  s0 = *(const uint4*)(sgn + i0),   s1 = *(const uint4*)(sgn + i0 + 4);
        uint4  s2 = *(const uint4*)(sgn + i1),   s3 = *(const uint4*)(sgn + i1 + 4);
        float4 h0 = *(const float4*)(h + i0),    h1 = *(const float4*)(h + i0 + 4);
        float4 h2 = *(const float4*)(h + i1),    h3 = *(const float4*)(h + i1 + 4);
        v16h a;
        a[0]  = (_Float16)g_elem(l0.x, s0.x, h0.x, useAgg, norm);
        a[1]  = (_Float16)g_elem(l0.y, s0.y, h0.y, useAgg, norm);
        a[2]  = (_Float16)g_elem(l0.z, s0.z, h0.z, useAgg, norm);
        a[3]  = (_Float16)g_elem(l0.w, s0.w, h0.w, useAgg, norm);
        a[4]  = (_Float16)g_elem(l1.x, s1.x, h1.x, useAgg, norm);
        a[5]  = (_Float16)g_elem(l1.y, s1.y, h1.y, useAgg, norm);
        a[6]  = (_Float16)g_elem(l1.z, s1.z, h1.z, useAgg, norm);
        a[7]  = (_Float16)g_elem(l1.w, s1.w, h1.w, useAgg, norm);
        a[8]  = (_Float16)g_elem(l2.x, s2.x, h2.x, useAgg, norm);
        a[9]  = (_Float16)g_elem(l2.y, s2.y, h2.y, useAgg, norm);
        a[10] = (_Float16)g_elem(l2.z, s2.z, h2.z, useAgg, norm);
        a[11] = (_Float16)g_elem(l2.w, s2.w, h2.w, useAgg, norm);
        a[12] = (_Float16)g_elem(l3.x, s3.x, h3.x, useAgg, norm);
        a[13] = (_Float16)g_elem(l3.y, s3.y, h3.y, useAgg, norm);
        a[14] = (_Float16)g_elem(l3.z, s3.z, h3.z, useAgg, norm);
        a[15] = (_Float16)g_elem(l3.w, s3.w, h3.w, useAgg, norm);

        #pragma unroll
        for (int nt = 0; nt < 8; ++nt) {
            v16h b = bf[(ks * 8 + nt) * 32 + lane];
            acc[nt] = __builtin_amdgcn_wmma_f32_16x16x32_f16(
                false, a, false, b, (short)0, acc[nt], false, false);
        }
    }
    int rbase = m0 + ((lane < 16) ? 0 : 8);
    #pragma unroll
    for (int nt = 0; nt < 8; ++nt)
        #pragma unroll
        for (int r = 0; r < 8; ++r)
            out[(size_t)(rbase + r) * OUT_F + nt * 16 + ncol] = acc[nt][r];
}

// ---------------------------------------------------------------------------
static inline size_t al256(size_t x) { return (x + 255) & ~(size_t)255; }

extern "C" void kernel_launch(void* const* d_in, const int* in_sizes, int n_in,
                              void* d_out, int out_size, void* d_ws, size_t ws_size,
                              hipStream_t stream) {
    const float* feat    = (const float*)d_in[0];
    const float* weight  = (const float*)d_in[1];
    const float* weight2 = (const float*)d_in[2];
    const float* ew      = (const float*)d_in[3];
    const int*   src     = (const int*)d_in[4];
    const int*   dst     = (const int*)d_in[5];
    float*       out     = (float*)d_out;

    const int nN = in_sizes[0] / IN_F;   // 100000 (multiple of 16)
    const int nE = in_sizes[3];          // 1600000
    const size_t NR = (size_t)nN * RANK;

    // Workspace carve-out (~78 MB)
    char* p = (char*)d_ws;
    size_t off = 0;
    int*       outdeg = (int*)      (p + off); off = al256(off + (size_t)nN * 4);
    int*       indeg  = (int*)      (p + off); off = al256(off + (size_t)nN * 4);
    float*     logs   = (float*)    (p + off); off = al256(off + NR * 4);
    unsigned*  sgn    = (unsigned*) (p + off); off = al256(off + NR * 4);
    float*     hbuf   = (float*)    (p + off); off = al256(off + NR * 4);
    _Float16*  w1frag = (_Float16*) (p + off); off = al256(off + (size_t)IN_F * RANK * 2);
    _Float16*  w2frag = (_Float16*) (p + off); off = al256(off + (size_t)RANK * OUT_F * 2);
    (void)ws_size; (void)n_in; (void)out_size;

    // 1) init / zero / pre-shuffle weights into fragment layout
    {
        int blocks = (int)((NR + 255) / 256);
        k_init<<<blocks, 256, 0, stream>>>(outdeg, indeg, logs, sgn,
                                           weight, weight2, w1frag, w2frag, nN);
    }
    // 2) degrees
    k_deg<<<(nE + 255) / 256, 256, 0, stream>>>(src, dst, outdeg, indeg, nE);

    // 3) GEMM1 + tanh (WMMA): one wave per 16-row tile
    {
        int mtiles = nN / 16;                        // 6250
        int blocks = (mtiles + 7) / 8;               // 8 waves / 256-thread block
        k_gemm1<<<blocks, 256, 0, stream>>>(feat, w1frag, outdeg, hbuf, mtiles);
    }
    // 4) edge scatter (log-space segment_prod), 4 ranks per thread
    {
        long long total = (long long)nE * (RANK / 4);   // 25.6M threads
        int blocks = (int)((total + 255) / 256);
        k_edge<<<blocks, 256, 0, stream>>>(hbuf, ew, src, dst, logs, sgn, total);
    }
    // 5) GEMM2 with fused finalize (WMMA) -> d_out
    {
        int mtiles = nN / 16;
        int blocks = (mtiles + 7) / 8;
        k_gemm2<<<blocks, 256, 0, stream>>>(logs, sgn, hbuf, indeg, w2frag, out, mtiles);
    }
}
